// MultiHeadAttention_61452392071691
// MI455X (gfx1250) — matrix operand order repaired and checked by reading, not compiled, not run on hardware
//
#include <hip/hip_runtime.h>
#include <hip/hip_bf16.h>
#include <stdint.h>

typedef __attribute__((ext_vector_type(16))) __bf16 v16bf;
typedef __attribute__((ext_vector_type(8)))  __bf16 v8bf;
typedef __attribute__((ext_vector_type(4)))  __bf16 v4bf;
typedef __attribute__((ext_vector_type(8)))  float  v8f;

#define EMB   1024
#define SEQ   2048
#define BATCH 4
#define HEADS 16
#define HDIM  64
#define BHTOT (BATCH*HEADS)

// ---------------- helpers ----------------

static __device__ __forceinline__ __bf16 f2bf(float f) {
  // round-to-nearest-even fp32 -> bf16
  unsigned u = __builtin_bit_cast(unsigned, f);
  unsigned r = u + 0x7FFFu + ((u >> 16) & 1u);
  unsigned short h = (unsigned short)(r >> 16);
  return __builtin_bit_cast(__bf16, h);
}

// A fragment, 16x32 bf16 (M x K). Lane L (L<16) row M=L holds
// K = kbase+0..15; lanes 16-31 hold K = kbase+16..31 (matches B loader).
// Two contiguous 16B loads per lane.
static __device__ __forceinline__ v16bf load_a16x32(const __bf16* base, int ld,
                                                    int kbase, int lane) {
  int m  = lane & 15;
  int ko = (lane >> 4) << 4;              // 0 or 16
  const __bf16* p = base + (size_t)m * ld + kbase + ko;
  union { v16bf v; v8bf h[2]; } u;
  u.h[0] = *(const v8bf*)p;
  u.h[1] = *(const v8bf*)(p + 8);
  return u.v;
}

// B fragment, 32x16 bf16 (K x N), sourced from a row-major [N,K] array so that
// B[k][n] = rowmajNK[n][k]. Lane n = lane&15 loads 16 contiguous K values
// starting at kbase + (lane>>4)*16 : a single 32B load.
static __device__ __forceinline__ v16bf load_b32x16(const __bf16* rowmajNK, int ld,
                                                    int ncol, int kbase, int lane) {
  int n  = lane & 15;
  int ko = (lane >> 4) << 4;              // 0 or 16
  const __bf16* p = rowmajNK + (size_t)(ncol + n) * ld + kbase + ko;
  return *(const v16bf*)p;
}

static __device__ __forceinline__ v8f wmma_bf16(v16bf a, v16bf b, v8f c) {
  return __builtin_amdgcn_wmma_f32_16x16x32_bf16(false, a, false, b,
                                                 (short)0, c, false, false);
}

// ---------------- kernels ----------------

// fp32 -> bf16, 4 elements per thread (n must be a multiple of 4; all our
// tensor sizes are).
__global__ void k_f32_to_bf16v4(const float* __restrict__ in,
                                __bf16* __restrict__ out, int n4) {
  int i = blockIdx.x * blockDim.x + threadIdx.x;
  if (i < n4) {
    float4 f = ((const float4*)in)[i];
    v4bf o = { f2bf(f.x), f2bf(f.y), f2bf(f.z), f2bf(f.w) };
    *(v4bf*)(out + (size_t)i * 4) = o;
  }
}

// QKV projection: C[8192,3072] = A[8192,1024] @ W[3072,1024]^T + bias.
// Wave tile 32x64 (2 M-tiles x 4 N-tiles): 8 WMMAs per k-step, B fragments
// reused across both M-tiles. Scatter epilogue into Q[bh,s,d], K[bh,s,d],
// Vt[bh,d,s] (bf16).
__global__ __launch_bounds__(256)
void k_gemm_qkv(const __bf16* __restrict__ A, const __bf16* __restrict__ W,
                const float* __restrict__ bias,
                __bf16* __restrict__ Qb, __bf16* __restrict__ Kb,
                __bf16* __restrict__ Vt) {
  const int lane = threadIdx.x & 31;
  const int wave = threadIdx.x >> 5;
  const int ncol = blockIdx.x * 64;
  const int mrow = blockIdx.y * 256 + wave * 32;
  const __bf16* Arow0 = A + (size_t)mrow * EMB;
  const __bf16* Arow1 = Arow0 + (size_t)16 * EMB;

  v8f acc[2][4] = {};
  for (int kb = 0; kb < EMB; kb += 32) {
    v16bf a0 = load_a16x32(Arow0, EMB, kb, lane);
    v16bf a1 = load_a16x32(Arow1, EMB, kb, lane);
#pragma unroll
    for (int j = 0; j < 4; ++j) {
      v16bf b = load_b32x16(W, EMB, ncol + j * 16, kb, lane);
      acc[0][j] = wmma_bf16(a0, b, acc[0][j]);
      acc[1][j] = wmma_bf16(a1, b, acc[1][j]);
    }
  }
  const int n_l = lane & 15;
  const int mg  = (lane >> 4) * 8;
#pragma unroll
  for (int j = 0; j < 4; ++j) {
    int c = ncol + j * 16 + n_l;
    float bv = bias[c];
    int which = c >> 10;          // 0=Q 1=K 2=V
    int e = c & 1023;
    int h = e >> 6, d = e & 63;
#pragma unroll
    for (int mi = 0; mi < 2; ++mi) {
#pragma unroll
      for (int i = 0; i < 8; ++i) {
        int r = mrow + mi * 16 + mg + i;
        int batch = r >> 11, srow = r & 2047;
        int bh = batch * HEADS + h;
        __bf16 val = f2bf(acc[mi][j][i] + bv);
        if (which == 0)      Qb[((size_t)bh * SEQ + srow) * HDIM + d] = val;
        else if (which == 1) Kb[((size_t)bh * SEQ + srow) * HDIM + d] = val;
        else                 Vt[((size_t)bh * HDIM + d) * SEQ + srow] = val;
      }
    }
  }
}

// Flash attention: one wave = 16 query rows, streams keys in chunks of 64.
__global__ __launch_bounds__(256)
void k_flash_attn(const __bf16* __restrict__ Q, const __bf16* __restrict__ K,
                  const __bf16* __restrict__ Vt, const uint8_t* __restrict__ mask,
                  __bf16* __restrict__ Ao) {
  __shared__ __attribute__((aligned(32))) __bf16 lds[8][16][64]; // wave-private P tiles
  const int lane = threadIdx.x & 31;
  const int wave = threadIdx.x >> 5;
  const int bh = blockIdx.x;
  const int batch = bh >> 4, head = bh & 15;
  const int q0 = blockIdx.y * 128 + wave * 16;

  const __bf16* Qh = Q  + (size_t)bh * SEQ * HDIM;
  const __bf16* Kh = K  + (size_t)bh * SEQ * HDIM;
  const __bf16* Vh = Vt + (size_t)bh * HDIM * SEQ;

  v16bf qa0 = load_a16x32(Qh + (size_t)q0 * HDIM, HDIM, 0, lane);
  v16bf qa1 = load_a16x32(Qh + (size_t)q0 * HDIM, HDIM, 32, lane);

  const int n_l = lane & 15;
  const int mg  = (lane >> 4) * 8;
  const float NEGBIG = -1.0e30f;
  float mrow[8], lrow[8];
#pragma unroll
  for (int i = 0; i < 8; ++i) { mrow[i] = NEGBIG; lrow[i] = 0.f; }
  v8f o[4] = {};
  const __bf16* Pt = &lds[wave][0][0];

  for (int kc = 0; kc < SEQ; kc += 64) {
    // S = (Q @ K^T) * 1/sqrt(d), 4 tiles of 16 keys
    v8f s[4];
#pragma unroll
    for (int j = 0; j < 4; ++j) {
      v16bf b0 = load_b32x16(Kh, HDIM, kc + j * 16, 0, lane);
      v16bf b1 = load_b32x16(Kh, HDIM, kc + j * 16, 32, lane);
      v8f t = {};
      t = wmma_bf16(qa0, b0, t);
      t = wmma_bf16(qa1, b1, t);
      s[j] = t;
    }
    // scale + mask
#pragma unroll
    for (int j = 0; j < 4; ++j) {
      int ki = kc + j * 16 + n_l;
#pragma unroll
      for (int i = 0; i < 8; ++i) {
        int qi = q0 + mg + i;
        float v = s[j][i] * 0.125f;
        if (!mask[(size_t)qi * SEQ + ki]) v = NEGBIG;
        s[j][i] = v;
      }
    }
    // online softmax: rows live across a 16-lane group, one element per lane.
#pragma unroll
    for (int i = 0; i < 8; ++i) {
      float mx = fmaxf(fmaxf(s[0][i], s[1][i]), fmaxf(s[2][i], s[3][i]));
#pragma unroll
      for (int off = 8; off > 0; off >>= 1) mx = fmaxf(mx, __shfl_xor(mx, off, 16));
      float mnew  = fmaxf(mrow[i], mx);
      float alpha = __expf(mrow[i] - mnew);
      float rs = 0.f;
#pragma unroll
      for (int j = 0; j < 4; ++j) {
        float p = __expf(s[j][i] - mnew);
        s[j][i] = p;
        rs += p;
      }
#pragma unroll
      for (int off = 8; off > 0; off >>= 1) rs += __shfl_xor(rs, off, 16);
      lrow[i] = lrow[i] * alpha + rs;
      mrow[i] = mnew;
#pragma unroll
      for (int j = 0; j < 4; ++j) o[j][i] *= alpha;
    }
    // C-layout -> A-fragment layout via wave-private LDS tile (DS is in-order
    // within a wave; no workgroup barrier needed).
#pragma unroll
    for (int j = 0; j < 4; ++j)
#pragma unroll
      for (int i = 0; i < 8; ++i)
        lds[wave][mg + i][j * 16 + n_l] = f2bf(s[j][i]);

    v16bf pa0 = load_a16x32(Pt, 64, 0, lane);
    v16bf pa1 = load_a16x32(Pt, 64, 32, lane);
#pragma unroll
    for (int j = 0; j < 4; ++j) {
      v16bf vb0 = load_b32x16(Vh, SEQ, j * 16, kc, lane);       // Vt rows contiguous in s
      v16bf vb1 = load_b32x16(Vh, SEQ, j * 16, kc + 32, lane);
      o[j] = wmma_bf16(pa0, vb0, o[j]);
      o[j] = wmma_bf16(pa1, vb1, o[j]);
    }
  }
  // normalize + write [b,s,h*64+d] bf16
#pragma unroll
  for (int i = 0; i < 8; ++i) lrow[i] = 1.f / lrow[i];
#pragma unroll
  for (int j = 0; j < 4; ++j)
#pragma unroll
    for (int i = 0; i < 8; ++i) {
      int r = batch * SEQ + q0 + mg + i;
      int c = head * HDIM + j * 16 + n_l;
      Ao[(size_t)r * EMB + c] = f2bf(o[j][i] * lrow[i]);
    }
}

// Output projection: out[8192,1024] = A[8192,1024] @ W[1024,1024]^T + bias.
// Same 32x64 wave tile as k_gemm_qkv; fp32 output in [b,s,E] layout.
__global__ __launch_bounds__(256)
void k_gemm_out(const __bf16* __restrict__ A, const __bf16* __restrict__ W,
                const float* __restrict__ bias, float* __restrict__ out) {
  const int lane = threadIdx.x & 31;
  const int wave = threadIdx.x >> 5;
  const int ncol = blockIdx.x * 64;
  const int mrow = blockIdx.y * 256 + wave * 32;
  const __bf16* Arow0 = A + (size_t)mrow * EMB;
  const __bf16* Arow1 = Arow0 + (size_t)16 * EMB;

  v8f acc[2][4] = {};
  for (int kb = 0; kb < EMB; kb += 32) {
    v16bf a0 = load_a16x32(Arow0, EMB, kb, lane);
    v16bf a1 = load_a16x32(Arow1, EMB, kb, lane);
#pragma unroll
    for (int j = 0; j < 4; ++j) {
      v16bf b = load_b32x16(W, EMB, ncol + j * 16, kb, lane);
      acc[0][j] = wmma_bf16(a0, b, acc[0][j]);
      acc[1][j] = wmma_bf16(a1, b, acc[1][j]);
    }
  }
  const int n_l = lane & 15, mg = (lane >> 4) * 8;
#pragma unroll
  for (int j = 0; j < 4; ++j) {
    float bv = bias[ncol + j * 16 + n_l];
#pragma unroll
    for (int mi = 0; mi < 2; ++mi)
#pragma unroll
      for (int i = 0; i < 8; ++i)
        out[(size_t)(mrow + mi * 16 + mg + i) * EMB + ncol + j * 16 + n_l] =
            acc[mi][j][i] + bv;
  }
}

// ---------------- launch ----------------

extern "C" void kernel_launch(void* const* d_in, const int* in_sizes, int n_in,
                              void* d_out, int out_size, void* d_ws, size_t ws_size,
                              hipStream_t stream) {
  (void)in_sizes; (void)n_in; (void)out_size; (void)ws_size;
  const float*   input = (const float*)d_in[0];
  const uint8_t* mask  = (const uint8_t*)d_in[1];
  const float*   qkv_w = (const float*)d_in[2];
  const float*   qkv_b = (const float*)d_in[3];
  const float*   out_w = (const float*)d_in[4];
  const float*   out_b = (const float*)d_in[5];
  float* out = (float*)d_out;

  char* ws = (char*)d_ws;
  size_t off = 0;
  auto alloc = [&](size_t bytes) -> char* {
    char* p = ws + off;
    off += (bytes + 255) & ~(size_t)255;
    return p;
  };
  __bf16* inA  = (__bf16*)alloc((size_t)BATCH * SEQ * EMB * 2);   // 16 MB
  __bf16* qkvw = (__bf16*)alloc((size_t)3 * EMB * EMB * 2);       //  6 MB
  __bf16* outw = (__bf16*)alloc((size_t)EMB * EMB * 2);           //  2 MB
  __bf16* Qb   = (__bf16*)alloc((size_t)BHTOT * SEQ * HDIM * 2);  // 16 MB
  __bf16* Kb   = (__bf16*)alloc((size_t)BHTOT * SEQ * HDIM * 2);  // 16 MB
  __bf16* Vt   = (__bf16*)alloc((size_t)BHTOT * HDIM * SEQ * 2);  // 16 MB
  __bf16* Ao   = (__bf16*)alloc((size_t)BATCH * SEQ * EMB * 2);   // 16 MB

  int n1 = BATCH * SEQ * EMB / 4;
  k_f32_to_bf16v4<<<(n1 + 255) / 256, 256, 0, stream>>>(input, inA, n1);
  int n2 = 3 * EMB * EMB / 4;
  k_f32_to_bf16v4<<<(n2 + 255) / 256, 256, 0, stream>>>(qkv_w, qkvw, n2);
  int n3 = EMB * EMB / 4;
  k_f32_to_bf16v4<<<(n3 + 255) / 256, 256, 0, stream>>>(out_w, outw, n3);

  k_gemm_qkv<<<dim3(3 * EMB / 64, BATCH * SEQ / 256), 256, 0, stream>>>(
      inA, qkvw, qkv_b, Qb, Kb, Vt);
  k_flash_attn<<<dim3(BHTOT, SEQ / 128), 256, 0, stream>>>(Qb, Kb, Vt, mask, Ao);
  k_gemm_out<<<dim3(EMB / 64, BATCH * SEQ / 256), 256, 0, stream>>>(
      Ao, outw, out_b, out);
}
